// SNAC_Lnet_1700807049927
// MI455X (gfx1250) — matrix operand order repaired and checked by reading, not compiled, not run on hardware
//
#include <hip/hip_runtime.h>
#include <math.h>

// ---------------------------------------------------------------------------
// Persistent fused LSTM+MLP rollout for MI455X (gfx1250, wave32, WMMA bf16).
//
// B=512, L=512, K=103, H=512. Padded input width KT=640:
//   A[b][0..102]   = x_t           (bf16)
//   A[b][103..104] = fed-back pos  (bf16)
//   A[b][105..111] = 0
//   A[b][112..623] = h_{t-1}       (bf16)
//   A[b][624..639] = 0
// Gate weights concatenated the same way: Wg[2048][640] (bf16).
//
// Block mapping: block b owns hidden tile nh = b>>2; its 8 waves cover batch
// tiles m = (b&3)*8 + wave. The 80KB gate-weight panel for nh (4 gates x 16
// rows x 640) is time-invariant -> preloaded into LDS once, WMMA B-fragments
// come from ds_load_b128 for all 512 steps (zero steady-state L2 B traffic).
// ---------------------------------------------------------------------------

#define Bsz   512
#define Lseq  512
#define Kobs  103
#define Hdim  512
#define KT    640
#define HOFF  112
#define NBLK  128
#define NTHR  256                      // 8 wave32 per workgroup
#define NWAVES (NBLK * (NTHR / 32))    // 1024 waves total
#define NGRP  (NBLK / 16)              // barrier groups (8)

// dynamic LDS partition (in u16 elements)
#define LB_ELEMS  (64 * KT)            // 64 x 640 weight panel
#define LY1_ELEMS (8 * 16 * 64)
#define LY2_ELEMS (8 * 16 * 16)
#define SMEM_BYTES ((LB_ELEMS + LY1_ELEMS + LY2_ELEMS) * 2)   // 102400 B

typedef __attribute__((ext_vector_type(16))) __bf16   v16bf;
typedef __attribute__((ext_vector_type(8)))  float    v8f;
typedef __attribute__((ext_vector_type(4)))  unsigned v4u;

union Frag {
  v16bf v;
  v4u   q[2];
  unsigned u[8];
};

// -------------------- persistent device state (re-inited every call) -------
__device__ unsigned short g_Wg[4 * Hdim * KT];   // 2048 x 640 bf16
__device__ unsigned short g_W1[64 * Hdim];       // 64 x 512 bf16
__device__ unsigned short g_W2[16 * 64];         // 16 x 64 bf16
__device__ unsigned short g_A[2][Bsz * KT];      // ping-pong activations
__device__ float          g_c[Bsz * Hdim];       // cell state f32
__device__ unsigned       g_sub[NGRP];           // barrier: group counters
__device__ unsigned       g_top;                 // barrier: top counter

// -------------------- helpers ---------------------------------------------
__device__ __forceinline__ unsigned short f2bf(float f) {
  unsigned u = __builtin_bit_cast(unsigned, f);
  return (unsigned short)((u + 0x7FFFu + ((u >> 16) & 1u)) >> 16); // RNE
}
__device__ __forceinline__ float bf2f(unsigned short h) {
  unsigned u = ((unsigned)h) << 16;
  return __builtin_bit_cast(float, u);
}
__device__ __forceinline__ float sigm(float x) { return 1.0f / (1.0f + __expf(-x)); }
__device__ __forceinline__ float tanh_(float x) {
  float e = __expf(2.0f * x);
  return 1.0f - 2.0f / (e + 1.0f);
}
__device__ __forceinline__ v8f vzero() { v8f z = {0,0,0,0,0,0,0,0}; return z; }

// A-fragment (16x32 bf16, row-major, stride in elements). ISA 7.12.2.
__device__ __forceinline__ v16bf load_a_frag(const unsigned short* base, int stride, int kbase) {
  const int lane = threadIdx.x & 31;
  const int M = lane & 15, half = lane >> 4;
  const unsigned short* row = base + M * stride + kbase + 8 * half;
  Frag f;
  f.q[0] = *(const v4u*)(row);        // K +0..7
  f.q[1] = *(const v4u*)(row + 16);   // K +16..23
  return f.v;
}

// B-fragment (32x16 bf16) from row-major W[N][K] (B = W^T).
__device__ __forceinline__ v16bf load_b_frag(const unsigned short* W, int stride, int n0, int kbase) {
  const int lane = threadIdx.x & 31;
  const int n = n0 + (lane & 15), half = lane >> 4;
  const unsigned short* row = W + n * stride + kbase + 8 * half;
  Frag f;
  f.q[0] = *(const v4u*)(row);        // K +0..7
  f.q[1] = *(const v4u*)(row + 16);   // K +16..23
  return f.v;
}

__device__ __forceinline__ v8f wmma_bf16(v16bf a, v16bf b, v8f c) {
  return __builtin_amdgcn_wmma_f32_16x16x32_bf16(false, a, false, b, (short)0, c, false, false);
}

// Two-level grid barrier, monotonic epochs: 16 blocks -> group ctr -> top ctr.
__device__ __forceinline__ void grid_bar(unsigned ep) {
  __syncthreads();
  if (threadIdx.x == 0) {
    __threadfence();
    const int grp = blockIdx.x >> 4;
    const unsigned old = atomicAdd(&g_sub[grp], 1u);
    if ((old & 15u) == 15u) atomicAdd(&g_top, 1u);   // 16th arrival forwards
    const unsigned tgt = ep * NGRP;
    while (*(volatile unsigned*)&g_top < tgt) __builtin_amdgcn_s_sleep(1);
    __threadfence();
  }
  __syncthreads();
}

// -------------------- prep kernels ----------------------------------------
__global__ void prep_weights(const float* __restrict__ W_ih, const float* __restrict__ W_hh,
                             const float* __restrict__ W1,   const float* __restrict__ W2) {
  const int stride = gridDim.x * blockDim.x;
  const int NWG = 4 * Hdim * KT, NW1 = 64 * Hdim, NW2 = 16 * 64;
  for (int e = blockIdx.x * blockDim.x + threadIdx.x; e < NWG + NW1 + NW2; e += stride) {
    if (e < NWG) {
      const int n = e / KT, k = e - n * KT;
      float v = 0.f;
      if (k < 105)                           v = W_ih[n * 105 + k];
      else if (k >= HOFF && k < HOFF + Hdim) v = W_hh[n * Hdim + (k - HOFF)];
      g_Wg[e] = f2bf(v);
    } else if (e < NWG + NW1) {
      g_W1[e - NWG] = f2bf(W1[e - NWG]);
    } else {
      g_W2[e - NWG - NW1] = f2bf(W2[e - NWG - NW1]);
    }
  }
}

__global__ void prep_state(const float* __restrict__ x, const float* __restrict__ pos,
                           const float* __restrict__ h0, const float* __restrict__ c0) {
  const int stride = gridDim.x * blockDim.x;
  const int idx0 = blockIdx.x * blockDim.x + threadIdx.x;
  if (idx0 == 0) g_top = 0u;
  if (idx0 < NGRP) g_sub[idx0] = 0u;
  const int NA = Bsz * KT;
  for (int e = idx0; e < 2 * NA + Bsz * Hdim; e += stride) {
    if (e < NA) {                         // A[0] for t=0
      const int b = e / KT, k = e - b * KT;
      float v = 0.f;
      if (k < Kobs)                          v = x[(size_t)b * Lseq * Kobs + k];
      else if (k < 105)                      v = pos[(size_t)b * Lseq * 2 + (k - Kobs)];
      else if (k >= HOFF && k < HOFF + Hdim) v = h0[b * Hdim + (k - HOFF)];
      g_A[0][e] = f2bf(v);
    } else if (e < 2 * NA) {
      g_A[1][e - NA] = 0;                 // zero (pads stay zero forever)
    } else {
      g_c[e - 2 * NA] = c0[e - 2 * NA];
    }
  }
}

// -------------------- persistent rollout kernel ----------------------------
__global__ __launch_bounds__(NTHR)
void lstm_mlp_persistent(const float* __restrict__ x,
                         const float* __restrict__ b_ih, const float* __restrict__ b_hh,
                         const float* __restrict__ b1,   const float* __restrict__ b2,
                         const float* __restrict__ W3,   const float* __restrict__ b3,
                         float* __restrict__ out) {
  extern __shared__ __align__(16) unsigned short smem[];
  unsigned short* lB  = smem;                   // [64][640] gate-weight panel
  unsigned short* ly1 = lB + LB_ELEMS;          // [8][16*64] y1 staging
  unsigned short* ly2 = ly1 + LY1_ELEMS;        // [8][16*16] y2 staging

  const int lane = threadIdx.x & 31;
  const int wv   = threadIdx.x >> 5;
  const int gw   = blockIdx.x * (NTHR / 32) + wv;     // 0..1023
  const int col  = lane & 15, half = lane >> 4;

  // block -> tile mapping: one hidden tile per block, 8 batch tiles (1/wave)
  const int nh = blockIdx.x >> 2;                     // 0..31
  const int m  = ((blockIdx.x & 3) << 3) + wv;        // 0..31
  const int n0 = nh * 16;
  const int hcol = n0 + col;

  // ---- one-time LDS preload of this block's gate-weight panel (80 KB) ----
  {
    const int r = threadIdx.x >> 2, q = threadIdx.x & 3;  // 64 rows x 4 quarters
    const int gate = r >> 4, nl = r & 15;
    const v4u* s = (const v4u*)(g_Wg + (size_t)(gate * Hdim + n0 + nl) * KT);
    v4u* d = (v4u*)(lB + r * KT);
    #pragma unroll
    for (int i = 0; i < 20; ++i) d[q * 20 + i] = s[q * 20 + i];
  }
  __syncthreads();

  // ---- loop-invariant per-lane values ----
  const float Bi = b_ih[hcol]            + b_hh[hcol];
  const float Bf = b_ih[Hdim + hcol]     + b_hh[Hdim + hcol];
  const float Bg = b_ih[2 * Hdim + hcol] + b_hh[2 * Hdim + hcol];
  const float Bo = b_ih[3 * Hdim + hcol] + b_hh[3 * Hdim + hcol];
  float bb1[4], w3r[16], bb2 = 0.f, bb3 = 0.f;
  if (gw < 32) {
    #pragma unroll
    for (int nt = 0; nt < 4; ++nt) bb1[nt] = b1[nt * 16 + col];
    bb2 = b2[col];
    bb3 = b3[half];
    #pragma unroll
    for (int k = 0; k < 16; ++k) w3r[k] = W3[half * 16 + k];
  }

  float* out_pos = out;                                // (B,L,2)
  float* out_h   = out + (size_t)Bsz * Lseq * 2;       // (B,H)
  float* out_c   = out_h + (size_t)Bsz * Hdim;         // (B,H)
  unsigned ep = 0;

  for (int t = 0; t < Lseq; ++t) {
    const unsigned short* Ap = g_A[t & 1];
    unsigned short*       An = g_A[(t + 1) & 1];

    // ================= Phase G: gates GEMM + LSTM cell =================
    {
      v8f ai = vzero(), af = vzero(), ag = vzero(), ao = vzero();
      const unsigned short* Ab = Ap + m * 16 * KT;
      #pragma unroll 4
      for (int kc = 0; kc < KT / 32; ++kc) {   // 20 K-chunks of 32
        const int kb = kc * 32;
        v16bf a = load_a_frag(Ab, KT, kb);     // global (L2-resident A)
        ai = wmma_bf16(a, load_b_frag(lB, KT,  0, kb), ai);   // LDS weights
        af = wmma_bf16(a, load_b_frag(lB, KT, 16, kb), af);
        ag = wmma_bf16(a, load_b_frag(lB, KT, 32, kb), ag);
        ao = wmma_bf16(a, load_b_frag(lB, KT, 48, kb), ao);
      }
      #pragma unroll
      for (int r = 0; r < 8; ++r) {
        const int row = m * 16 + r + 8 * half;        // C-tile: lanes16-31 = M+8
        const float iv = sigm(ai[r] + Bi);
        const float fv = sigm(af[r] + Bf);
        const float gv = tanh_(ag[r] + Bg);
        const float ov = sigm(ao[r] + Bo);
        const float cold = g_c[row * Hdim + hcol];
        const float cnew = fv * cold + iv * gv;
        g_c[row * Hdim + hcol] = cnew;
        const float hv = ov * tanh_(cnew);
        An[row * KT + HOFF + hcol] = f2bf(hv);        // h for next step
        if (t == Lseq - 1) {
          out_h[row * Hdim + hcol] = hv;
          out_c[row * Hdim + hcol] = cnew;
        }
      }
    }
    grid_bar(++ep);

    // ============ Phase M: MLP head (waves 0..31) / x staging ============
    if (gw < 32) {
      const int mm = gw;
      // layer1: (16 x 512) @ (512 -> 64)
      v8f acc[4];
      acc[0] = acc[1] = acc[2] = acc[3] = vzero();
      const unsigned short* Hb = An + mm * 16 * KT + HOFF;
      #pragma unroll 2
      for (int kc = 0; kc < Hdim / 32; ++kc) {
        const int kb = kc * 32;
        v16bf a = load_a_frag(Hb, KT, kb);
        #pragma unroll
        for (int nt = 0; nt < 4; ++nt)
          acc[nt] = wmma_bf16(a, load_b_frag(g_W1, Hdim, nt * 16, kb), acc[nt]);
      }
      unsigned short* w_ly1 = ly1 + wv * (16 * 64);
      unsigned short* w_ly2 = ly2 + wv * (16 * 16);
      #pragma unroll
      for (int nt = 0; nt < 4; ++nt) {
        #pragma unroll
        for (int r = 0; r < 8; ++r) {
          float v = fmaxf(acc[nt][r] + bb1[nt], 0.f);
          w_ly1[(r + 8 * half) * 64 + nt * 16 + col] = f2bf(v);
        }
      }
      // layer2: (16 x 64) @ (64 -> 16) — re-fragment via LDS, same wave
      v8f a2 = vzero();
      #pragma unroll
      for (int kc = 0; kc < 2; ++kc) {
        const int kb = kc * 32;
        v16bf a = load_a_frag(w_ly1, 64, kb);
        a2 = wmma_bf16(a, load_b_frag(g_W2, 64, 0, kb), a2);
      }
      #pragma unroll
      for (int r = 0; r < 8; ++r) {
        float v = fmaxf(a2[r] + bb2, 0.f);
        w_ly2[(r + 8 * half) * 16 + col] = f2bf(v);
      }
      // layer3: 16x2 per tile; lane -> (row=col, j=half)
      {
        float s = bb3;
        #pragma unroll
        for (int k = 0; k < 16; ++k)
          s += bf2f(w_ly2[col * 16 + k]) * w3r[k];
        s = fmaxf(s, 0.f);
        const int brow = mm * 16 + col;
        out_pos[(size_t)brow * (Lseq * 2) + t * 2 + half] = s;
        An[brow * KT + 103 + half] = f2bf(s);          // pos feedback
      }
    } else if (t + 1 < Lseq) {
      // 992 waves stage x[:, t+1, :] into the next A buffer (bf16)
      const int tid2 = (gw - 32) * 32 + lane;
      const int stride2 = (NWAVES - 32) * 32;
      for (int e = tid2; e < Bsz * Kobs; e += stride2) {
        const int row = e / Kobs, cc = e - row * Kobs;
        const size_t xbase = (size_t)row * Lseq * Kobs + cc;
        An[row * KT + cc] = f2bf(x[xbase + (size_t)(t + 1) * Kobs]);
        if (t + 2 < Lseq)
          __builtin_prefetch(&x[xbase + (size_t)(t + 2) * Kobs], 0, 1);
      }
    }
    grid_bar(++ep);
  }
}

// -------------------- host entry -------------------------------------------
extern "C" void kernel_launch(void* const* d_in, const int* in_sizes, int n_in,
                              void* d_out, int out_size, void* d_ws, size_t ws_size,
                              hipStream_t stream) {
  (void)in_sizes; (void)n_in; (void)out_size; (void)d_ws; (void)ws_size;
  const float* x    = (const float*)d_in[0];
  const float* pos  = (const float*)d_in[1];
  const float* h0   = (const float*)d_in[2];
  const float* c0   = (const float*)d_in[3];
  const float* W_ih = (const float*)d_in[4];
  const float* W_hh = (const float*)d_in[5];
  const float* b_ih = (const float*)d_in[6];
  const float* b_hh = (const float*)d_in[7];
  const float* W1   = (const float*)d_in[8];
  const float* b1   = (const float*)d_in[9];
  const float* W2   = (const float*)d_in[10];
  const float* b2   = (const float*)d_in[11];
  const float* W3   = (const float*)d_in[12];
  const float* b3   = (const float*)d_in[13];
  float* out = (float*)d_out;

  hipLaunchKernelGGL(prep_weights, dim3(1024), dim3(256), 0, stream, W_ih, W_hh, W1, W2);
  hipLaunchKernelGGL(prep_state,   dim3(512),  dim3(256), 0, stream, x, pos, h0, c0);
  hipLaunchKernelGGL(lstm_mlp_persistent, dim3(NBLK), dim3(NTHR), SMEM_BYTES, stream,
                     x, b_ih, b_hh, b1, b2, W3, b3, out);
}
